// SwinTransformerBlock_11450382811310
// MI455X (gfx1250) — matrix operand order repaired and checked by reading, not compiled, not run on hardware
//
#include <hip/hip_runtime.h>
#include <hip/hip_bf16.h>

// ---------------------------------------------------------------------------
// Swin block for MI455X / gfx1250 (wave32, WMMA bf16 -> f32 accumulate)
// Shapes: B=32 C=256 H=W=56 WS=7 SS=3 HEADS=8 hd=32 HID=1024
// Token rows R = 32*64*49 = 100352 (divisible by 128 -> edge-free GEMM tiling)
// ---------------------------------------------------------------------------

#define RTOK   100352
#define CDIM   256
#define HIDDIM 1024
#define NHEADS 8

typedef unsigned short u16;
typedef __attribute__((ext_vector_type(16))) __bf16 v16bf;
typedef __attribute__((ext_vector_type(8)))  float  v8f;

union Frag {
  v16bf v;
  u16   s[16];
  uint4 q[2];
};

__device__ __forceinline__ u16 f32_to_bf16(float f) {
  union { float f; unsigned int u; } c; c.f = f;
  unsigned int u = c.u;
  u += 0x7FFFu + ((u >> 16) & 1u);      // round-to-nearest-even
  return (u16)(u >> 16);
}
__device__ __forceinline__ float bf16_to_f32(u16 h) {
  union { unsigned int u; float f; } c; c.u = ((unsigned int)h) << 16;
  return c.f;
}

__device__ __forceinline__ v8f wmma_bf16(const Frag& a, const Frag& b, v8f c) {
  return __builtin_amdgcn_wmma_f32_16x16x32_bf16(
      false, a.v, false, b.v, (short)0, c, false, false);
}

// ---------------------------------------------------------------------------
// K0: fp32 -> bf16 weight conversion (weights then stay L2-resident)
// ---------------------------------------------------------------------------
__global__ void f2bf_kernel(const float* __restrict__ src, u16* __restrict__ dst, int n) {
  int i = blockIdx.x * blockDim.x + threadIdx.x;
  if (i < n) dst[i] = f32_to_bf16(src[i]);
}

// ---------------------------------------------------------------------------
// K1: window-partition (with -3 roll) + LayerNorm1.
// One 256-thread block per token row; thread = channel.
// Writes: shortcut (f32, R x 256) and LN output (bf16, R x 256).
// ---------------------------------------------------------------------------
__global__ __launch_bounds__(256)
void win_ln1_kernel(const float* __restrict__ x,
                    const float* __restrict__ gw, const float* __restrict__ gb,
                    float* __restrict__ shortcut, u16* __restrict__ lnout) {
  int r = blockIdx.x;
  int c = threadIdx.x;
  int win = r / 49, t = r % 49;
  int b = win >> 6, wloc = win & 63;
  int wh = wloc >> 3, ww = wloc & 7;
  int i = t / 7, j = t % 7;
  int h0 = wh * 7 + i + 3; if (h0 >= 56) h0 -= 56;
  int w0 = ww * 7 + j + 3; if (w0 >= 56) w0 -= 56;
  float val = x[(((long)b * CDIM + c) * 56 + h0) * 56 + w0];

  __shared__ float redA[8], redB[8];
  float s = val, q = val * val;
  for (int off = 16; off > 0; off >>= 1) {
    s += __shfl_down(s, off, 32);
    q += __shfl_down(q, off, 32);
  }
  int lane = threadIdx.x & 31, wv = threadIdx.x >> 5;
  if (lane == 0) { redA[wv] = s; redB[wv] = q; }
  __syncthreads();
  if (threadIdx.x == 0) {
    float ts = 0.f, tq = 0.f;
    for (int k = 0; k < 8; ++k) { ts += redA[k]; tq += redB[k]; }
    redA[0] = ts; redB[0] = tq;
  }
  __syncthreads();
  float mean = redA[0] * (1.f / 256.f);
  float var  = redB[0] * (1.f / 256.f) - mean * mean;
  float rstd = rsqrtf(var + 1e-5f);

  shortcut[(long)r * CDIM + c] = val;
  float ln = (val - mean) * rstd * gw[c] + gb[c];
  lnout[(long)r * CDIM + c] = f32_to_bf16(ln);
}

// ---------------------------------------------------------------------------
// K4: plain row LayerNorm (for norm2), f32 in -> bf16 out
// ---------------------------------------------------------------------------
__global__ __launch_bounds__(256)
void ln_rows_kernel(const float* __restrict__ y,
                    const float* __restrict__ gw, const float* __restrict__ gb,
                    u16* __restrict__ lnout) {
  int r = blockIdx.x;
  int c = threadIdx.x;
  float val = y[(long)r * CDIM + c];

  __shared__ float redA[8], redB[8];
  float s = val, q = val * val;
  for (int off = 16; off > 0; off >>= 1) {
    s += __shfl_down(s, off, 32);
    q += __shfl_down(q, off, 32);
  }
  int lane = threadIdx.x & 31, wv = threadIdx.x >> 5;
  if (lane == 0) { redA[wv] = s; redB[wv] = q; }
  __syncthreads();
  if (threadIdx.x == 0) {
    float ts = 0.f, tq = 0.f;
    for (int k = 0; k < 8; ++k) { ts += redA[k]; tq += redB[k]; }
    redA[0] = ts; redB[0] = tq;
  }
  __syncthreads();
  float mean = redA[0] * (1.f / 256.f);
  float var  = redB[0] * (1.f / 256.f) - mean * mean;
  float rstd = rsqrtf(var + 1e-5f);
  float ln = (val - mean) * rstd * gw[c] + gb[c];
  lnout[(long)r * CDIM + c] = f32_to_bf16(ln);
}

// ---------------------------------------------------------------------------
// K2: GEMM  C[r,n] = sum_k A[r,k] * W[n,k]   (A: R x K bf16, W: N x K bf16)
// Block: 256 threads = 8 waves; tile 128(M) x 64(N); wave = 32x32 (2x2 WMMA).
// K%32==0, M%128==0, N%64==0 for every call -> no edge handling.
// Epilogue modes:
//   0: out_bf16 = acc + bias                         (qkv)
//   1: resid   += acc + bias   (f32, in-place)       (proj + residual)
//   2: out_bf16 = gelu(acc + bias)                   (fc1)
//   3: d_out[scatter] = resid + acc + bias           (fc2 + residual + unwindow)
// ---------------------------------------------------------------------------
__global__ __launch_bounds__(256)
void gemm_wmma_kernel(const u16* __restrict__ A, const u16* __restrict__ W,
                      const float* __restrict__ bias,
                      float* __restrict__ resid,
                      u16* __restrict__ out_bf16,
                      float* __restrict__ out_f32,
                      int K, int N, int mode) {
  int lane = threadIdx.x & 31;
  int wv   = threadIdx.x >> 5;
  int mw   = wv & 3;           // 0..3 -> which 32-row group
  int nw   = wv >> 2;          // 0..1 -> which 32-col group
  int half = lane >> 4;        // 0/1
  int mn   = lane & 15;

  long blockM = (long)blockIdx.x * 128;
  int  blockN = blockIdx.y * 64;

  long rowA0 = blockM + mw * 32 + mn;     // lane's A row, m-subtile 0
  long rowA1 = rowA0 + 16;                // m-subtile 1
  long colB0 = blockN + nw * 32 + mn;     // lane's W row (output col), n-sub 0
  long colB1 = colB0 + 16;

  v8f acc00 = {}, acc01 = {}, acc10 = {}, acc11 = {};

  for (int k0 = 0; k0 < K; k0 += 32) {
    Frag a0, a1, b0, b1;
    // A layout (16x32 bf16): lane -> row; elems e at K=k0+half*16+e (contig 32B)
    const u16* pa0 = A + rowA0 * K + k0 + half * 16;
    a0.q[0] = *(const uint4*)(pa0);
    a0.q[1] = *(const uint4*)(pa0 + 8);
    const u16* pa1 = A + rowA1 * K + k0 + half * 16;
    a1.q[0] = *(const uint4*)(pa1);
    a1.q[1] = *(const uint4*)(pa1 + 8);
    // B layout (32x16 bf16): lane -> col; elems e at K=k0+half*16+e (contig 32B)
    const u16* pb0 = W + colB0 * K + k0 + half * 16;
    b0.q[0] = *(const uint4*)(pb0);
    b0.q[1] = *(const uint4*)(pb0 + 8);
    const u16* pb1 = W + colB1 * K + k0 + half * 16;
    b1.q[0] = *(const uint4*)(pb1);
    b1.q[1] = *(const uint4*)(pb1 + 8);

    acc00 = wmma_bf16(a0, b0, acc00);
    acc01 = wmma_bf16(a0, b1, acc01);
    acc10 = wmma_bf16(a1, b0, acc10);
    acc11 = wmma_bf16(a1, b1, acc11);
  }

  v8f accs[2][2] = {{acc00, acc01}, {acc10, acc11}};
  for (int mi = 0; mi < 2; ++mi) {
    for (int ni = 0; ni < 2; ++ni) {
      for (int rr = 0; rr < 8; ++rr) {
        long row = blockM + mw * 32 + mi * 16 + rr + half * 8;
        int  col = blockN + nw * 32 + ni * 16 + mn;
        float v = accs[mi][ni][rr];
        if (mode == 0) {
          out_bf16[row * N + col] = f32_to_bf16(v + bias[col]);
        } else if (mode == 1) {
          long idx = row * CDIM + col;
          resid[idx] = resid[idx] + v + bias[col];
        } else if (mode == 2) {
          float t = v + bias[col];
          float g = 0.5f * t * (1.0f + erff(t * 0.70710678118654752f));
          out_bf16[row * N + col] = f32_to_bf16(g);
        } else {
          float t = resid[row * CDIM + col] + v + bias[col];
          // un-window + roll(+3) scatter back to (B, C, H, W)
          int r32 = (int)row;
          int win = r32 / 49, tk = r32 % 49;
          int b = win >> 6, wloc = win & 63;
          int wh = wloc >> 3, ww = wloc & 7;
          int ii = tk / 7, jj = tk % 7;
          int h0 = wh * 7 + ii + 3; if (h0 >= 56) h0 -= 56;
          int w0 = ww * 7 + jj + 3; if (w0 >= 56) w0 -= 56;
          out_f32[(((long)b * CDIM + col) * 56 + h0) * 56 + w0] = t;
        }
      }
    }
  }
}

// ---------------------------------------------------------------------------
// K3: windowed attention, one wave (32 threads) per (window, head).
// S = (Q*scale) Kt + rpb + shiftmask ; softmax ; O = P V.  N padded 49 -> 64.
// qkv buffer: (R x 768) bf16, cols [0:256)=Q [256:512)=K [512:768)=V,
// within each: head*32 + d.
// ---------------------------------------------------------------------------
__device__ __forceinline__ int region9(int p, int q) {
  int rh = (p < 49) ? 0 : ((p < 53) ? 1 : 2);
  int rw = (q < 49) ? 0 : ((q < 53) ? 1 : 2);
  return rh * 3 + rw;
}

__global__ __launch_bounds__(32)
void attn_kernel(const u16* __restrict__ qkv, const float* __restrict__ rpb,
                 u16* __restrict__ out) {
  int win  = blockIdx.x;            // 0..2047
  int h    = blockIdx.y;            // 0..7
  int lane = threadIdx.x;
  int half = lane >> 4, mn = lane & 15;
  int wloc = win & 63;
  int wh = wloc >> 3, ww = wloc & 7;
  long base = (long)win * 49 * 768;

  // ---- Q as A-fragments (4 M-tiles, K = head_dim = 32 in one WMMA) ----
  Frag qa[4];
  for (int ti = 0; ti < 4; ++ti) {
    int row = ti * 16 + mn;
    if (row < 49) {
      const u16* p = qkv + base + (long)row * 768 + h * 32 + half * 16;
      qa[ti].q[0] = *(const uint4*)(p);
      qa[ti].q[1] = *(const uint4*)(p + 8);
    } else {
      qa[ti].q[0] = uint4{0, 0, 0, 0};
      qa[ti].q[1] = uint4{0, 0, 0, 0};
    }
  }
  // ---- K as B-fragments (4 N-tiles): B[k,n] = K[n,k], contiguous in k ----
  Frag kb[4];
  for (int tj = 0; tj < 4; ++tj) {
    int col = tj * 16 + mn;
    if (col < 49) {
      const u16* p = qkv + base + (long)col * 768 + 256 + h * 32 + half * 16;
      kb[tj].q[0] = *(const uint4*)(p);
      kb[tj].q[1] = *(const uint4*)(p + 8);
    } else {
      kb[tj].q[0] = uint4{0, 0, 0, 0};
      kb[tj].q[1] = uint4{0, 0, 0, 0};
    }
  }

  // ---- scores ----
  v8f s[4][4] = {};
  for (int ti = 0; ti < 4; ++ti)
    for (int tj = 0; tj < 4; ++tj)
      s[ti][tj] = wmma_bf16(qa[ti], kb[tj], s[ti][tj]);

  // ---- scale + rel-pos bias + shift mask + column masking ----
  const float scale = 0.17677669529663687f;   // 1/sqrt(32)
  for (int tj = 0; tj < 4; ++tj) {
    int col = tj * 16 + mn;
    bool colValid = col < 49;
    int ib = 0, jb = 0, labC = 0;
    if (colValid) {
      ib = col / 7; jb = col % 7;
      labC = region9(wh * 7 + ib, ww * 7 + jb);
    }
    for (int ti = 0; ti < 4; ++ti) {
      for (int rr = 0; rr < 8; ++rr) {
        int row = ti * 16 + rr + half * 8;
        float v = s[ti][tj][rr] * scale;
        if (!colValid) {
          v = -1e30f;
        } else if (row < 49) {
          int ia = row / 7, ja = row % 7;
          int idx = (ia - ib + 6) * 13 + (ja - jb + 6);
          v += rpb[idx * NHEADS + h];
          int labR = region9(wh * 7 + ia, ww * 7 + ja);
          if (labR != labC) v -= 100.0f;
        }
        s[ti][tj][rr] = v;
      }
    }
  }

  // ---- row softmax: rows striped over 16-lane halves; reduce via shfl_xor --
  for (int ti = 0; ti < 4; ++ti) {
    for (int rr = 0; rr < 8; ++rr) {
      float m = s[ti][0][rr];
      m = fmaxf(m, s[ti][1][rr]);
      m = fmaxf(m, s[ti][2][rr]);
      m = fmaxf(m, s[ti][3][rr]);
      for (int msk = 1; msk < 16; msk <<= 1)
        m = fmaxf(m, __shfl_xor(m, msk, 32));
      float sum = 0.f;
      for (int tj = 0; tj < 4; ++tj) {
        float e = __expf(s[ti][tj][rr] - m);
        s[ti][tj][rr] = e;
        sum += e;
      }
      for (int msk = 1; msk < 16; msk <<= 1)
        sum += __shfl_xor(sum, msk, 32);
      float inv = 1.0f / sum;
      for (int tj = 0; tj < 4; ++tj) s[ti][tj][rr] *= inv;
    }
  }

  // ---- restripe P through LDS into A-fragment layout ----
  __shared__ u16 p_lds[64 * 64];
  for (int ti = 0; ti < 4; ++ti)
    for (int tj = 0; tj < 4; ++tj)
      for (int rr = 0; rr < 8; ++rr)
        p_lds[(ti * 16 + rr + half * 8) * 64 + tj * 16 + mn] =
            f32_to_bf16(s[ti][tj][rr]);
  __syncthreads();

  // ---- O = P @ V : K=64 in two 32-steps; V gathered (zero-padded rows) ----
  v8f o[4][2] = {};
  for (int ks = 0; ks < 2; ++ks) {
    Frag pa[4];
    for (int ti = 0; ti < 4; ++ti) {
      const u16* p = &p_lds[(ti * 16 + mn) * 64 + ks * 32 + half * 16];
      pa[ti].q[0] = *(const uint4*)(p);
      pa[ti].q[1] = *(const uint4*)(p + 8);
    }
    Frag vb[2];
    for (int td = 0; td < 2; ++td) {
      int d = td * 16 + mn;
      for (int e = 0; e < 16; ++e) {
        int k = ks * 32 + half * 16 + e;
        vb[td].s[e] = (k < 49)
            ? qkv[base + (long)k * 768 + 512 + h * 32 + d]
            : (u16)0;
      }
    }
    for (int ti = 0; ti < 4; ++ti)
      for (int td = 0; td < 2; ++td)
        o[ti][td] = wmma_bf16(pa[ti], vb[td], o[ti][td]);
  }

  // ---- store (rows < 49 only) ----
  for (int ti = 0; ti < 4; ++ti)
    for (int td = 0; td < 2; ++td)
      for (int rr = 0; rr < 8; ++rr) {
        int row = ti * 16 + rr + half * 8;
        if (row < 49)
          out[((long)win * 49 + row) * CDIM + h * 32 + td * 16 + mn] =
              f32_to_bf16(o[ti][td][rr]);
      }
}

// ---------------------------------------------------------------------------
extern "C" void kernel_launch(void* const* d_in, const int* in_sizes, int n_in,
                              void* d_out, int out_size, void* d_ws, size_t ws_size,
                              hipStream_t stream) {
  const float* x      = (const float*)d_in[0];
  const float* n1w    = (const float*)d_in[1];
  const float* n1b    = (const float*)d_in[2];
  const float* qkv_w  = (const float*)d_in[3];
  const float* qkv_b  = (const float*)d_in[4];
  const float* rpb    = (const float*)d_in[5];
  const float* proj_w = (const float*)d_in[6];
  const float* proj_b = (const float*)d_in[7];
  const float* n2w    = (const float*)d_in[8];
  const float* n2b    = (const float*)d_in[9];
  const float* fc1_w  = (const float*)d_in[10];
  const float* fc1_b  = (const float*)d_in[11];
  const float* fc2_w  = (const float*)d_in[12];
  const float* fc2_b  = (const float*)d_in[13];
  float* out = (float*)d_out;

  char* ws = (char*)d_ws;
  size_t off = 0;
  float* shortcut = (float*)(ws + off); off += (size_t)RTOK * CDIM * 4;   // f32 residual
  u16*   bufA     = (u16*)(ws + off);   off += (size_t)RTOK * CDIM * 2;   // LN outputs
  u16*   bufB     = (u16*)(ws + off);   off += (size_t)RTOK * HIDDIM * 2; // qkv / fc1 out
  u16*   bufC     = (u16*)(ws + off);   off += (size_t)RTOK * CDIM * 2;   // attn out
  u16*   wq  = (u16*)(ws + off); off += (size_t)768 * 256 * 2;
  u16*   wp  = (u16*)(ws + off); off += (size_t)256 * 256 * 2;
  u16*   wf1 = (u16*)(ws + off); off += (size_t)1024 * 256 * 2;
  u16*   wf2 = (u16*)(ws + off); off += (size_t)256 * 1024 * 2;

  // K0: weights -> bf16 (stay resident in 192MB L2 thereafter)
  f2bf_kernel<<<(768 * 256 + 255) / 256, 256, 0, stream>>>(qkv_w, wq, 768 * 256);
  f2bf_kernel<<<(256 * 256 + 255) / 256, 256, 0, stream>>>(proj_w, wp, 256 * 256);
  f2bf_kernel<<<(1024 * 256 + 255) / 256, 256, 0, stream>>>(fc1_w, wf1, 1024 * 256);
  f2bf_kernel<<<(256 * 1024 + 255) / 256, 256, 0, stream>>>(fc2_w, wf2, 256 * 1024);

  // K1: shifted window partition + LN1
  win_ln1_kernel<<<RTOK, 256, 0, stream>>>(x, n1w, n1b, shortcut, bufA);

  // qkv GEMM: (R x 256) x (768 x 256)^T -> bf16
  gemm_wmma_kernel<<<dim3(RTOK / 128, 768 / 64), 256, 0, stream>>>(
      bufA, wq, qkv_b, nullptr, bufB, nullptr, 256, 768, 0);

  // attention: one wave per (window, head)
  attn_kernel<<<dim3(2048, NHEADS), 32, 0, stream>>>(bufB, rpb, bufC);

  // proj GEMM + residual (in-place into shortcut, f32)
  gemm_wmma_kernel<<<dim3(RTOK / 128, 256 / 64), 256, 0, stream>>>(
      bufC, wp, proj_b, shortcut, nullptr, nullptr, 256, 256, 1);

  // LN2
  ln_rows_kernel<<<RTOK, 256, 0, stream>>>(shortcut, n2w, n2b, bufA);

  // fc1 GEMM + exact GELU -> bf16
  gemm_wmma_kernel<<<dim3(RTOK / 128, 1024 / 64), 256, 0, stream>>>(
      bufA, wf1, fc1_b, nullptr, bufB, nullptr, 256, 1024, 2);

  // fc2 GEMM + residual + un-window/roll scatter -> d_out (f32 NCHW)
  gemm_wmma_kernel<<<dim3(RTOK / 128, 256 / 64), 256, 0, stream>>>(
      bufB, wf2, fc2_b, shortcut, nullptr, out, 1024, 256, 3);
}
